// deepar_cedar_38946763440819
// MI455X (gfx1250) — matrix operand order repaired and checked by reading, not compiled, not run on hardware
//
#include <hip/hip_runtime.h>
#include <hip/hip_bf16.h>
#include <math.h>

#define T_DIM 192
#define B_DIM 256
#define D_HID 128
#define G_DIM 512            // 4*D_HID
#define L_OUT 24             // d_outputseqlen (reference constant)
#define D_REPF (L_OUT * 3 * D_HID)   // 9216
#define N_DOM 32
#define EPS_C 1e-6f
#define LOG_2PI_C 1.8378770664093453f

typedef __bf16 v16bf __attribute__((ext_vector_type(16)));
typedef float  v8f   __attribute__((ext_vector_type(8)));
typedef unsigned int v4u  __attribute__((ext_vector_type(4)));
typedef int          v4i  __attribute__((ext_vector_type(4)));
typedef int          v8i  __attribute__((ext_vector_type(8)));

union Frag16 { v16bf v; v4u q[2]; };

// LSTM dynamic-LDS layout: [ Whh bf16 128KB | gates f32 32KB | hbf bf16 4KB ]
#define W_LDS_BYTES   (G_DIM * D_HID * 2)              // 131072
#define GATES_OFF     W_LDS_BYTES
#define HBF_OFF       (GATES_OFF + 16 * G_DIM * 4)     // +32768
#define LSTM_LDS_SZ   (HBF_OFF + 16 * D_HID * 2)       // 167936

__device__ __forceinline__ float sigmoidf_(float x) { return 1.0f / (1.0f + __expf(-x)); }
__device__ __forceinline__ float softplusf_(float x) {
  return x > 0.f ? x + log1pf(__expf(-x)) : log1pf(__expf(x));
}

// ---------------------------------------------------------------------------
// 1) Build LSTM-0 input matrix (T*B, 64) in bf16: [lag(4) | cov(8) | emb0(20)
//    | emb1(10) | zero-pad to 64].
// ---------------------------------------------------------------------------
__global__ void k_build_inputs(const int* __restrict__ Xe, const float* __restrict__ Xc,
                               const float* __restrict__ Xl,
                               const float* __restrict__ emb0, const float* __restrict__ emb1,
                               __bf16* __restrict__ out) {
  int idx = blockIdx.x * blockDim.x + threadIdx.x;
  const int total = T_DIM * B_DIM * 64;
  if (idx >= total) return;
  int col = idx & 63;
  int row = idx >> 6;
  float v = 0.f;
  if (col < 4)       v = Xl[row * 4 + col];
  else if (col < 12) v = Xc[row * 8 + (col - 4)];
  else if (col < 32) v = emb0[Xe[row * 2 + 0] * 20 + (col - 12)];
  else if (col < 42) v = emb1[Xe[row * 2 + 1] * 10 + (col - 32)];
  out[idx] = (__bf16)v;
}

__global__ void k_w2bf(const float* __restrict__ W, __bf16* __restrict__ out,
                       int rows, int cs, int cd) {
  int idx = blockIdx.x * blockDim.x + threadIdx.x;
  if (idx >= rows * cd) return;
  int r = idx / cd, c = idx % cd;
  out[idx] = (c < cs) ? (__bf16)W[r * cs + c] : (__bf16)0.f;
}

__global__ void k_f2bf(const float* __restrict__ x, __bf16* __restrict__ out, int n) {
  int idx = blockIdx.x * blockDim.x + threadIdx.x;
  if (idx < n) out[idx] = (__bf16)x[idx];
}

// ---------------------------------------------------------------------------
// 2) WMMA GEMM: xg[M,512] = X[M,Kpad](bf16) @ W[512,Kpad]^T (bf16) + b_ih+b_hh
//    CDNA5 wave32 fragment layouts:
//      A lane L: row = L&15; K-runs [kt+(L>>4)*8, +8) and [kt+16+(L>>4)*8, +8)
//      B lane L: col = n0 + (L&15); K-runs [kt+(L>>4)*8, +8) and [kt+16+(L>>4)*8, +8)
//      D lane L: rows (L>>4)*8 .. +7, col = n0 + (L&15)
// ---------------------------------------------------------------------------
__global__ __launch_bounds__(256) void k_gemm_xg(const __bf16* __restrict__ X, int Kpad,
                                                 const __bf16* __restrict__ W,
                                                 const float* __restrict__ bih,
                                                 const float* __restrict__ bhh,
                                                 float* __restrict__ xg) {
  const int lane = threadIdx.x & 31;
  const int wid  = threadIdx.x >> 5;
  const int m0   = (blockIdx.x * 8 + wid) * 16;
  const int half = lane >> 4;
  const int lcol = lane & 15;
  const __bf16* arow = X + (size_t)(m0 + lcol) * Kpad;

  for (int nc = 0; nc < 8; ++nc) {
    v8f acc[4];
    for (int j = 0; j < 4; ++j)
      for (int r = 0; r < 8; ++r) acc[j][r] = 0.f;

    for (int kt = 0; kt < Kpad; kt += 32) {
      Frag16 a;
      const __bf16* ap = arow + kt + half * 8;
      a.q[0] = *(const v4u*)(ap);
      a.q[1] = *(const v4u*)(ap + 16);
      for (int j = 0; j < 4; ++j) {
        int ncol = nc * 64 + j * 16 + lcol;
        const __bf16* bp = W + (size_t)ncol * Kpad + kt + half * 8;
        Frag16 b;
        b.q[0] = *(const v4u*)(bp);
        b.q[1] = *(const v4u*)(bp + 16);
        acc[j] = __builtin_amdgcn_wmma_f32_16x16x32_bf16(
            false, a.v, false, b.v, (short)0, acc[j], false, false);
      }
    }
    for (int j = 0; j < 4; ++j) {
      int ncol = nc * 64 + j * 16 + lcol;
      float bias = bih[ncol] + bhh[ncol];
      for (int r = 0; r < 8; ++r) {
        int row = m0 + half * 8 + r;
        xg[(size_t)row * G_DIM + ncol] = acc[j][r] + bias;
      }
    }
  }
}

// ---------------------------------------------------------------------------
// 3) LSTM recurrence. One block (128 thr = 4 waves) per 16-row batch tile.
//    W_hh (512x128 bf16, 128 KB) is staged into LDS once via the Tensor Data
//    Mover (tensor_load_to_lds, TENSORcnt), so the 192-step serial loop reads
//    B-fragments from LDS only. Gate tiles via chained WMMAs (K=128 -> 4).
// ---------------------------------------------------------------------------
__global__ __launch_bounds__(128) void k_lstm(const float* __restrict__ xg,
                                              const __bf16* __restrict__ Whh,
                                              float* __restrict__ hout) {
  extern __shared__ char smem[];
  __bf16* Wlds = (__bf16*)smem;                    // LDS offset 0 (dynamic base)
  float*  gates = (float*)(smem + GATES_OFF);
  __bf16* hbf   = (__bf16*)(smem + HBF_OFF);

  const int tid  = threadIdx.x;
  const int lane = tid & 31;
  const int wid  = tid >> 5;
  const int half = lane >> 4;
  const int lcol = lane & 15;
  const int b0   = blockIdx.x * 16;

  // --- TDM: DMA Whh (2D tile 128 x 512 rows, bf16) into LDS offset 0 -------
  if (wid == 0) {
    unsigned long long ga = (unsigned long long)Whh;
    v4u g0;
    g0[0] = 1u;                                       // count=1, user descriptor
    g0[1] = 0u;                                       // lds_addr = 0
    g0[2] = (unsigned)(ga & 0xFFFFFFFFull);           // global_addr[31:0]
    g0[3] = (unsigned)((ga >> 32) & 0x01FFFFFFull) | (2u << 30);  // addr[56:32] | type=2
    v8i g1;
    g1[0] = 0x00010000;                               // data_size=1 (2 bytes)
    g1[1] = (int)(128u << 16);                        // tensor_dim0 = 128 (bits 79:48)
    g1[2] = (int)(512u << 16);                        // tensor_dim1 = 512 (bits 111:80)
    g1[3] = (int)(128u << 16);                        // tile_dim0 = 128 (bits 127:112)
    g1[4] = 512;                                      // tile_dim1 = 512, tile_dim2 = 0
    g1[5] = 128;                                      // tensor_dim0_stride = 128
    g1[6] = 0;
    g1[7] = 0;
    v4i gz = {0, 0, 0, 0};
#if __clang_major__ >= 23
    __builtin_amdgcn_tensor_load_to_lds(g0, g1, gz, gz, (v8i)0, 0);
#else
    __builtin_amdgcn_tensor_load_to_lds(g0, g1, gz, gz, 0);
#endif
    __builtin_amdgcn_s_wait_tensorcnt(0);
  }

  float creg[16];
  for (int m = 0; m < 16; ++m) creg[m] = 0.f;
  for (int i = tid; i < 16 * D_HID; i += 128) hbf[i] = (__bf16)0.f;
  __syncthreads();

  for (int t = 0; t < T_DIM; ++t) {
    const float* xgt = xg + ((size_t)t * B_DIM + b0) * G_DIM;
    v8f acc[8];
    for (int i = 0; i < 8; ++i) {
      int ncol = (wid * 8 + i) * 16 + lcol;
      for (int r = 0; r < 8; ++r)
        acc[i][r] = xgt[(size_t)(half * 8 + r) * G_DIM + ncol];
    }
    for (int kt = 0; kt < D_HID; kt += 32) {
      Frag16 a;
      const __bf16* ap = hbf + lcol * D_HID + kt + half * 8;
      a.q[0] = *(const v4u*)(ap);
      a.q[1] = *(const v4u*)(ap + 16);
      for (int i = 0; i < 8; ++i) {
        int ncol = (wid * 8 + i) * 16 + lcol;
        const __bf16* bp = Wlds + (size_t)ncol * D_HID + kt + half * 8;
        Frag16 b;
        b.q[0] = *(const v4u*)(bp);
        b.q[1] = *(const v4u*)(bp + 16);
        acc[i] = __builtin_amdgcn_wmma_f32_16x16x32_bf16(
            false, a.v, false, b.v, (short)0, acc[i], false, false);
      }
    }
    for (int i = 0; i < 8; ++i) {
      int ncol = (wid * 8 + i) * 16 + lcol;
      for (int r = 0; r < 8; ++r)
        gates[(half * 8 + r) * G_DIM + ncol] = acc[i][r];
    }
    __syncthreads();
    {
      const int j = tid;  // hidden-unit column owned by this thread
      for (int m = 0; m < 16; ++m) {
        float gi = gates[m * G_DIM + j];
        float gf = gates[m * G_DIM + 128 + j];
        float gg = gates[m * G_DIM + 256 + j];
        float go = gates[m * G_DIM + 384 + j];
        float c = sigmoidf_(gf) * creg[m] + sigmoidf_(gi) * tanhf(gg);
        float h = sigmoidf_(go) * tanhf(c);
        creg[m] = c;
        hout[((size_t)t * B_DIM + b0 + m) * D_HID + j] = h;
        hbf[m * D_HID + j] = (__bf16)h;
      }
    }
    __syncthreads();
  }
}

// ---------------------------------------------------------------------------
// 4) Head: per-batch thread computes loc/scale/logp over last L steps.
// ---------------------------------------------------------------------------
__global__ void k_head(const float* __restrict__ h0, const float* __restrict__ h1,
                       const float* __restrict__ h2, const float* __restrict__ y,
                       const float* __restrict__ locW, const float* __restrict__ locB,
                       const float* __restrict__ scW, const float* __restrict__ scB,
                       float* __restrict__ yloc, float* __restrict__ yscale,
                       float* __restrict__ lb) {
  int b = blockIdx.x * blockDim.x + threadIdx.x;
  if (b >= B_DIM) return;
  float s = 0.f;
  for (int t = 0; t < L_OUT; ++t) {
    int tg = T_DIM - L_OUT + t;
    size_t roff = ((size_t)tg * B_DIM + b) * D_HID;
    float aloc = locB[0], asc = scB[0];
    for (int j = 0; j < D_HID; ++j) {
      float a = h0[roff + j], bb = h1[roff + j], cc = h2[roff + j];
      aloc += a * locW[j] + bb * locW[128 + j] + cc * locW[256 + j];
      asc  += a * scW[j]  + bb * scW[128 + j]  + cc * scW[256 + j];
    }
    float scale = softplusf_(asc) + EPS_C;
    float yt = y[(size_t)tg * B_DIM + b];
    float z = (yt - aloc) / scale;
    float nlp = 0.5f * z * z + logf(scale) + 0.5f * LOG_2PI_C;  // = -logp
    yloc[t * B_DIM + b] = aloc;
    yscale[t * B_DIM + b] = scale;
    s += nlp;
  }
  lb[b] = s / (float)L_OUT;
}

// ---------------------------------------------------------------------------
// 5) dom_rep[d, f]: deterministic fixed-order segment mean over batch.
// ---------------------------------------------------------------------------
__global__ void k_domrep(const float* __restrict__ h0, const float* __restrict__ h1,
                         const float* __restrict__ h2, const int* __restrict__ dom,
                         float* __restrict__ dom_rep) {
  int idx = blockIdx.x * blockDim.x + threadIdx.x;
  if (idx >= N_DOM * D_REPF) return;
  int d = idx / D_REPF;
  int f = idx % D_REPF;
  int t = f / (3 * D_HID);
  int rem = f % (3 * D_HID);
  int layer = rem / D_HID;
  int j = rem % D_HID;
  const float* h = (layer == 0) ? h0 : ((layer == 1) ? h1 : h2);
  int tg = T_DIM - L_OUT + t;
  float sum = 0.f; int cnt = 0;
  for (int b = 0; b < B_DIM; ++b) {
    if (dom[b] == d) { sum += h[((size_t)tg * B_DIM + b) * D_HID + j]; ++cnt; }
  }
  dom_rep[idx] = sum / (float)(cnt > 0 ? cnt : 1);
}

// ---------------------------------------------------------------------------
// 6) Finalize: loss_y, dom_loss, MMD penalty, total loss (deterministic).
// ---------------------------------------------------------------------------
__global__ __launch_bounds__(256) void k_finalize(const float* __restrict__ dom_rep,
                                                  const float* __restrict__ lb,
                                                  const int* __restrict__ dom,
                                                  float* __restrict__ out) {
  __shared__ float red[256];
  __shared__ float sq[N_DOM];
  __shared__ float dl[N_DOM];
  const int tid = threadIdx.x;

  red[tid] = lb[tid];
  __syncthreads();
  for (int s = 128; s > 0; s >>= 1) {
    if (tid < s) red[tid] += red[tid + s];
    __syncthreads();
  }
  float loss_y = red[0] / (float)B_DIM;
  __syncthreads();

  if (tid < N_DOM) {
    float s = 0.f; int c = 0;
    for (int b = 0; b < B_DIM; ++b)
      if (dom[b] == tid) { s += lb[b]; ++c; }
    dl[tid] = s / (float)(c > 0 ? c : 1);
    float q = 0.f;
    const float* r = dom_rep + (size_t)tid * D_REPF;
    for (int f = 0; f < D_REPF; ++f) q += r[f] * r[f];
    sq[tid] = q;
  }
  __syncthreads();

  float pen = 0.f;
  for (int p = tid; p < N_DOM * N_DOM; p += 256) {
    int d = p / N_DOM, e = p % N_DOM;
    const float* rd = dom_rep + (size_t)d * D_REPF;
    const float* re = dom_rep + (size_t)e * D_REPF;
    float dot = 0.f;
    for (int f = 0; f < D_REPF; ++f) dot += rd[f] * re[f];
    float mmd = sq[d] + sq[e] - 2.f * dot;
    pen += mmd * fabsf(dl[d] - dl[e]);
  }
  red[tid] = pen;
  __syncthreads();
  for (int s = 128; s > 0; s >>= 1) {
    if (tid < s) red[tid] += red[tid + s];
    __syncthreads();
  }
  if (tid == 0) {
    float penalty = red[0] / (float)((N_DOM - 1) * N_DOM);   // GAMMA = 1
    out[0] = loss_y + penalty;
    out[1] = loss_y;
  }
}

// ---------------------------------------------------------------------------
extern "C" void kernel_launch(void* const* d_in, const int* in_sizes, int n_in,
                              void* d_out, int out_size, void* d_ws, size_t ws_size,
                              hipStream_t stream) {
  const int*   Xe   = (const int*)d_in[0];
  const float* Xc   = (const float*)d_in[1];
  const float* Xl   = (const float*)d_in[2];
  const float* y    = (const float*)d_in[3];
  const int*   dom  = (const int*)d_in[4];
  // d_in[5] = d_outputseqlen == 24 (compile-time constant L_OUT)
  const float* emb0 = (const float*)d_in[6];
  const float* emb1 = (const float*)d_in[7];
  const float* Wih[3] = {(const float*)d_in[8],  (const float*)d_in[12], (const float*)d_in[16]};
  const float* Whh[3] = {(const float*)d_in[9],  (const float*)d_in[13], (const float*)d_in[17]};
  const float* bih[3] = {(const float*)d_in[10], (const float*)d_in[14], (const float*)d_in[18]};
  const float* bhh[3] = {(const float*)d_in[11], (const float*)d_in[15], (const float*)d_in[19]};
  const float* locW = (const float*)d_in[20];
  const float* locB = (const float*)d_in[21];
  const float* scW  = (const float*)d_in[22];
  const float* scB  = (const float*)d_in[23];

  char* ws = (char*)d_ws;
  size_t off = 0;
  auto alloc = [&](size_t bytes) -> void* {
    void* p = ws + off;
    off = (off + bytes + 255) & ~(size_t)255;
    return p;
  };
  const int M = T_DIM * B_DIM;  // 49152

  __bf16* inputs_bf = (__bf16*)alloc((size_t)M * 64 * sizeof(__bf16));
  __bf16* Wihb[3];
  __bf16* Whhb[3];
  Wihb[0] = (__bf16*)alloc((size_t)G_DIM * 64 * sizeof(__bf16));   // padded K 42->64
  Whhb[0] = (__bf16*)alloc((size_t)G_DIM * D_HID * sizeof(__bf16));
  Wihb[1] = (__bf16*)alloc((size_t)G_DIM * D_HID * sizeof(__bf16));
  Whhb[1] = (__bf16*)alloc((size_t)G_DIM * D_HID * sizeof(__bf16));
  Wihb[2] = (__bf16*)alloc((size_t)G_DIM * D_HID * sizeof(__bf16));
  Whhb[2] = (__bf16*)alloc((size_t)G_DIM * D_HID * sizeof(__bf16));
  __bf16* hXbf = (__bf16*)alloc((size_t)M * D_HID * sizeof(__bf16));
  float*  xg   = (float*)alloc((size_t)M * G_DIM * sizeof(float));
  float*  h[3];
  for (int l = 0; l < 3; ++l) h[l] = (float*)alloc((size_t)M * D_HID * sizeof(float));
  float* lb      = (float*)alloc(B_DIM * sizeof(float));
  float* dom_rep = (float*)alloc((size_t)N_DOM * D_REPF * sizeof(float));

  // LSTM kernel needs 164 KB dynamic LDS (CDNA5 WGP has 320 KB)
  hipFuncSetAttribute((const void*)k_lstm,
                      hipFuncAttributeMaxDynamicSharedMemorySize, (int)LSTM_LDS_SZ);

  // 1) gather/concat inputs (bf16, K padded to 64)
  { int n = M * 64; k_build_inputs<<<(n + 255) / 256, 256, 0, stream>>>(Xe, Xc, Xl, emb0, emb1, inputs_bf); }

  // 2) convert weights to bf16 (layer-0 W_ih zero-padded 42 -> 64)
  { int n = G_DIM * 64;    k_w2bf<<<(n + 255) / 256, 256, 0, stream>>>(Wih[0], Wihb[0], G_DIM, 42, 64); }
  { int n = G_DIM * D_HID; k_w2bf<<<(n + 255) / 256, 256, 0, stream>>>(Whh[0], Whhb[0], G_DIM, D_HID, D_HID); }
  { int n = G_DIM * D_HID; k_w2bf<<<(n + 255) / 256, 256, 0, stream>>>(Wih[1], Wihb[1], G_DIM, D_HID, D_HID); }
  { int n = G_DIM * D_HID; k_w2bf<<<(n + 255) / 256, 256, 0, stream>>>(Whh[1], Whhb[1], G_DIM, D_HID, D_HID); }
  { int n = G_DIM * D_HID; k_w2bf<<<(n + 255) / 256, 256, 0, stream>>>(Wih[2], Wihb[2], G_DIM, D_HID, D_HID); }
  { int n = G_DIM * D_HID; k_w2bf<<<(n + 255) / 256, 256, 0, stream>>>(Whh[2], Whhb[2], G_DIM, D_HID, D_HID); }

  // 3) layer 0: xg = inputs @ Wih0^T + b ; then recurrence
  k_gemm_xg<<<M / 128, 256, 0, stream>>>(inputs_bf, 64, Wihb[0], bih[0], bhh[0], xg);
  k_lstm<<<B_DIM / 16, 128, LSTM_LDS_SZ, stream>>>(xg, Whhb[0], h[0]);

  // 4) layers 1..2
  for (int l = 1; l < 3; ++l) {
    int n = M * D_HID;
    k_f2bf<<<(n + 255) / 256, 256, 0, stream>>>(h[l - 1], hXbf, n);
    k_gemm_xg<<<M / 128, 256, 0, stream>>>(hXbf, D_HID, Wihb[l], bih[l], bhh[l], xg);
    k_lstm<<<B_DIM / 16, 128, LSTM_LDS_SZ, stream>>>(xg, Whhb[l], h[l]);
  }

  // 5) heads + losses
  float* out = (float*)d_out;
  k_head<<<4, 64, 0, stream>>>(h[0], h[1], h[2], y, locW, locB, scW, scB,
                               out + 2, out + 2 + L_OUT * B_DIM, lb);
  { int n = N_DOM * D_REPF; k_domrep<<<(n + 255) / 256, 256, 0, stream>>>(h[0], h[1], h[2], dom, dom_rep); }
  k_finalize<<<1, 256, 0, stream>>>(dom_rep, lb, dom, out);

  (void)in_sizes; (void)n_in; (void)out_size; (void)ws_size;
}
